// RayTracing_21062519620204
// MI455X (gfx1250) — matrix operand order repaired and checked by reading, not compiled, not run on hardware
//
#include <hip/hip_runtime.h>
#include <hip/hip_bf16.h>

typedef _Float16 v16h __attribute__((ext_vector_type(16)));
typedef _Float16 h2v  __attribute__((ext_vector_type(2)));
typedef float    v8f  __attribute__((ext_vector_type(8)));

#define HDIM 256
#define LSTRIDE 260           // halfs per LDS row (pad vs 256 to spread banks)
#define NSTEPS 100
#define NSECANT 8
#define SPHERE_ITERS 10
#define SDF_THRESH 5e-5f

// ---- dynamic LDS layout (byte offsets) ----
#define OFF_W2T   0
#define SZ_W2T    (HDIM * LSTRIDE * 2)            // 133120
#define OFF_HA    (OFF_W2T + SZ_W2T)
#define SZ_HA     (32 * LSTRIDE * 2)              // 16640
#define OFF_W1    (OFF_HA + SZ_HA)                // 768 f32
#define OFF_B1    (OFF_W1 + 768 * 4)              // 256 f32
#define OFF_B2    (OFF_B1 + 256 * 4)              // 256 f32
#define OFF_W3    (OFF_B2 + 256 * 4)              // 256 f32
#define OFF_B3    (OFF_W3 + 256 * 4)              // 1 f32 (pad 16)
#define OFF_PTS   (OFF_B3 + 16)                   // 32*3 f32
#define OFF_BASE  (OFF_PTS + 32 * 3 * 4)          // 32 f32
#define OFF_SDF   (OFF_BASE + 32 * 4)             // 32 f32
#define OFF_PART  (OFF_SDF + 32 * 4)              // 32*8 f32
#define OFF_CAM   (OFF_PART + 32 * 8 * 4)         // 48 f32
#define OFF_RAY   (OFF_CAM + 48 * 4)              // 48 f32
#define LDS_TOTAL (OFF_RAY + 48 * 4)              // = 157968 bytes

// Batched SDF eval: pts[32][3] (LDS) -> sdfv[32] (LDS). All 256 threads enter.
__device__ __forceinline__ void sdf_eval(char* smem, int tid) {
  _Float16* W2T  = (_Float16*)(smem + OFF_W2T);
  _Float16* hA   = (_Float16*)(smem + OFF_HA);
  float* w1   = (float*)(smem + OFF_W1);
  float* b1   = (float*)(smem + OFF_B1);
  float* b2   = (float*)(smem + OFF_B2);
  float* w3   = (float*)(smem + OFF_W3);
  float* b3   = (float*)(smem + OFF_B3);
  float* pts  = (float*)(smem + OFF_PTS);
  float* basev= (float*)(smem + OFF_BASE);
  float* sdfv = (float*)(smem + OFF_SDF);
  float* part = (float*)(smem + OFF_PART);

  __syncthreads();                       // pts written by caller
  // ---- layer 1: h = tanh(p @ W1 + b1), 8 threads per point, 32 cols each
  {
    int p = tid >> 3, cb = (tid & 7) * 32;
    float px = pts[p * 3 + 0], py = pts[p * 3 + 1], pz = pts[p * 3 + 2];
    if ((tid & 7) == 0)
      basev[p] = sqrtf(px * px + py * py + pz * pz + 1e-12f) - 0.6f;
    for (int c = cb; c < cb + 32; ++c) {
      float a = b1[c] + px * w1[c] + py * w1[256 + c] + pz * w1[512 + c];
      hA[p * LSTRIDE + c] = (_Float16)tanhf(a);
    }
  }
  __syncthreads();
  // ---- layer 2: D(32x256) = hA(32x256) x W2(256x256) + b2, via WMMA f16
  int lane = tid & 31;
  int wv   = tid >> 5;        // wave id 0..7
  int half = lane >> 4;       // lane half
  int MN   = lane & 15;       // M for A-frag / N for B,C,D frags
  v8f accs[4];
  #pragma unroll
  for (int t = 0; t < 4; ++t) {
    int tile = wv + 8 * t;               // 0..31 over (2 Mtiles x 16 Ntiles)
    int mt = tile >> 4, nt = tile & 15;
    float bcol = b2[nt * 16 + MN];
    v8f c;
    #pragma unroll
    for (int r = 0; r < 8; ++r) c[r] = bcol;
    #pragma unroll
    for (int kk = 0; kk < 8; ++kk) {     // K = 256 in steps of 32
      v16h a, b;
      #pragma unroll
      for (int v = 0; v < 8; ++v) {
        // A layout (ISA 7.12.2, 16-bit A 16x32): K = 2*(v&3) + 8*half + 16*(v>>2)
        int ka = ((v & 3) * 2) + half * 8 + ((v >> 2) * 16) + kk * 32;
        h2v pa = *(const h2v*)&hA[(mt * 16 + MN) * LSTRIDE + ka];
        a[2 * v]     = pa.x;
        a[2 * v + 1] = pa.y;
        // B layout (32x16): same K order as A: K = 2*(v&3) + 8*half + 16*(v>>2)
        int kb = ((v & 3) * 2) + half * 8 + ((v >> 2) * 16) + kk * 32;
        h2v pb = *(const h2v*)&W2T[(nt * 16 + MN) * LSTRIDE + kb];
        b[2 * v]     = pb.x;
        b[2 * v + 1] = pb.y;
      }
      c = __builtin_amdgcn_wmma_f32_16x16x32_f16(false, a, false, b,
                                                 (short)0, c, false, false);
    }
    accs[t] = c;
  }
  __syncthreads();                        // all waves done reading hA
  // tanh + in-place writeback as f16 (D row = r + 8*half, col = MN)
  #pragma unroll
  for (int t = 0; t < 4; ++t) {
    int tile = wv + 8 * t;
    int mt = tile >> 4, nt = tile & 15;
    #pragma unroll
    for (int r = 0; r < 8; ++r) {
      int row = mt * 16 + r + 8 * half;
      hA[row * LSTRIDE + nt * 16 + MN] = (_Float16)tanhf(accs[t][r]);
    }
  }
  __syncthreads();
  // ---- layer 3: sdf = base + 0.05*(h2 @ W3 + b3)
  {
    int p = tid >> 3, cb = (tid & 7) * 32;
    float s = 0.f;
    for (int c = cb; c < cb + 32; ++c)
      s += (float)hA[p * LSTRIDE + c] * w3[c];
    part[p * 8 + (tid & 7)] = s;
  }
  __syncthreads();
  if (tid < 32) {
    float s = 0.f;
    #pragma unroll
    for (int j = 0; j < 8; ++j) s += part[tid * 8 + j];
    sdfv[tid] = basev[tid] + 0.05f * (s + b3[0]);
  }
  __syncthreads();
}

__global__ void __launch_bounds__(256)
raytrace_kernel(const float* __restrict__ camg, const float* __restrict__ rayg,
                const float* __restrict__ W1g, const float* __restrict__ b1g,
                const float* __restrict__ W2g, const float* __restrict__ b2g,
                const float* __restrict__ W3g, const float* __restrict__ b3g,
                float* __restrict__ out, int R) {
  extern __shared__ char smem[];
  int tid = threadIdx.x;
  int r0  = blockIdx.x * 16;

  _Float16* W2T = (_Float16*)(smem + OFF_W2T);
  float* w1   = (float*)(smem + OFF_W1);
  float* b1   = (float*)(smem + OFF_B1);
  float* b2   = (float*)(smem + OFF_B2);
  float* w3   = (float*)(smem + OFF_W3);
  float* b3   = (float*)(smem + OFF_B3);
  float* pts  = (float*)(smem + OFF_PTS);
  float* sdfv = (float*)(smem + OFF_SDF);
  float* camL = (float*)(smem + OFF_CAM);
  float* rayL = (float*)(smem + OFF_RAY);

  // ---- stage weights: W2 (f32, k-major) -> W2^T in LDS as f16 [n][k]
  for (int e = tid; e < HDIM * HDIM; e += 256) {
    int k = e >> 8, n = e & 255;
    W2T[n * LSTRIDE + k] = (_Float16)W2g[e];
  }
  for (int e = tid; e < 768; e += 256) w1[e] = W1g[e];
  if (tid < 256) { b1[tid] = b1g[tid]; b2[tid] = b2g[tid]; w3[tid] = W3g[tid]; }
  if (tid == 0) b3[0] = b3g[0];
  if (tid < 48) {
    int g = r0 * 3 + tid;
    int gc = (g < R * 3) ? g : (R * 3 - 1);
    camL[tid] = camg[gc];
    rayL[tid] = rayg[gc];
  }
  __syncthreads();

  // ---- per-ray state lives in registers of threads 0..15
  int ri = tid;
  float cx = 0, cy = 0, cz = 0, rx = 0, ry = 0, rz = 1;
  bool mask_int = false, unf_s = false, unf_e = false;
  float acc_s = 0, acc_e = 0, nsdf_s = 0, nsdf_e = 0;
  if (ri < 16) {
    cx = camL[ri * 3 + 0]; cy = camL[ri * 3 + 1]; cz = camL[ri * 3 + 2];
    rx = rayL[ri * 3 + 0]; ry = rayL[ri * 3 + 1]; rz = rayL[ri * 3 + 2];
    float rcd   = rx * cx + ry * cy + rz * cz;
    float under = rcd * rcd - (cx * cx + cy * cy + cz * cz - 1.0f);
    mask_int = under > 0.f;
    float sq = sqrtf(mask_int ? under : 1.0f);
    float i0 = mask_int ? fmaxf(-sq - rcd, 0.f) : 0.f;
    float i1 = mask_int ? fmaxf( sq - rcd, 0.f) : 0.f;
    unf_s = mask_int; unf_e = mask_int;
    acc_s = i0; acc_e = i1;
    pts[ri * 3 + 0] = cx + acc_s * rx; pts[ri * 3 + 1] = cy + acc_s * ry; pts[ri * 3 + 2] = cz + acc_s * rz;
    pts[(16 + ri) * 3 + 0] = cx + acc_e * rx; pts[(16 + ri) * 3 + 1] = cy + acc_e * ry; pts[(16 + ri) * 3 + 2] = cz + acc_e * rz;
  }
  sdf_eval(smem, tid);
  if (ri < 16) {
    nsdf_s = unf_s ? sdfv[ri] : 0.f;
    nsdf_e = unf_e ? sdfv[16 + ri] : 0.f;
  }

  // ---- sphere tracing from both ends
  #pragma unroll 1
  for (int it = 0; it < SPHERE_ITERS; ++it) {
    float csdf_s = 0, csdf_e = 0; bool np_s = false, np_e = false;
    if (ri < 16) {
      csdf_s = unf_s ? nsdf_s : 0.f; if (csdf_s <= SDF_THRESH) csdf_s = 0.f;
      csdf_e = unf_e ? nsdf_e : 0.f; if (csdf_e <= SDF_THRESH) csdf_e = 0.f;
      unf_s = unf_s && (csdf_s > SDF_THRESH);
      unf_e = unf_e && (csdf_e > SDF_THRESH);
      acc_s += csdf_s; acc_e -= csdf_e;
      pts[ri * 3 + 0] = cx + acc_s * rx; pts[ri * 3 + 1] = cy + acc_s * ry; pts[ri * 3 + 2] = cz + acc_s * rz;
      pts[(16 + ri) * 3 + 0] = cx + acc_e * rx; pts[(16 + ri) * 3 + 1] = cy + acc_e * ry; pts[(16 + ri) * 3 + 2] = cz + acc_e * rz;
    }
    sdf_eval(smem, tid);
    if (ri < 16) {
      nsdf_s = unf_s ? sdfv[ri] : 0.f;
      nsdf_e = unf_e ? sdfv[16 + ri] : 0.f;
      np_s = nsdf_s < 0.f; np_e = nsdf_e < 0.f;
      if (np_s) acc_s -= 0.5f * csdf_s;
      if (np_e) acc_e += 0.5f * csdf_e;
      pts[ri * 3 + 0] = cx + acc_s * rx; pts[ri * 3 + 1] = cy + acc_s * ry; pts[ri * 3 + 2] = cz + acc_s * rz;
      pts[(16 + ri) * 3 + 0] = cx + acc_e * rx; pts[(16 + ri) * 3 + 1] = cy + acc_e * ry; pts[(16 + ri) * 3 + 2] = cz + acc_e * rz;
    }
    sdf_eval(smem, tid);
    if (ri < 16) {
      if (np_s) nsdf_s = sdfv[ri];
      if (np_e) nsdf_e = sdfv[16 + ri];
      bool lt = acc_s < acc_e;
      unf_s = unf_s && lt; unf_e = unf_e && lt;
    }
  }

  bool sampler_mask = false, net_obj = false;
  if (ri < 16) {
    float cs = unf_s ? nsdf_s : 0.f; if (cs <= SDF_THRESH) cs = 0.f;
    unf_s = unf_s && (cs > SDF_THRESH);
    sampler_mask = unf_s;
    net_obj = acc_s < acc_e;
  }

  // ---- uniform sampler, streaming argmin of sign(sdf)*(N-s)
  float smin = 0, smax = 0;
  float bestval = 1e30f; int ind = 0;
  float z_ind = 0, s_ind = 0, zlo = 0, slo = 0, zps = 0, sps = 0;
  if (ri < 16) { smin = sampler_mask ? acc_s : 0.f; smax = sampler_mask ? acc_e : 0.f; }
  #pragma unroll 1
  for (int bb = 0; bb < NSTEPS / 2; ++bb) {
    int s0 = 2 * bb;
    float z0 = 0, z1 = 0;
    if (ri < 16) {
      z0 = smin + (smax - smin) * ((float)s0 / (float)(NSTEPS - 1));
      z1 = smin + (smax - smin) * ((float)(s0 + 1) / (float)(NSTEPS - 1));
      pts[ri * 3 + 0] = cx + z0 * rx; pts[ri * 3 + 1] = cy + z0 * ry; pts[ri * 3 + 2] = cz + z0 * rz;
      pts[(16 + ri) * 3 + 0] = cx + z1 * rx; pts[(16 + ri) * 3 + 1] = cy + z1 * ry; pts[(16 + ri) * 3 + 2] = cz + z1 * rz;
    }
    sdf_eval(smem, tid);
    if (ri < 16) {
      #pragma unroll
      for (int u = 0; u < 2; ++u) {
        int s = s0 + u;
        float z  = u ? z1 : z0;
        float sv = sdfv[u * 16 + ri];
        float sg = (sv > 0.f) ? 1.f : ((sv < 0.f) ? -1.f : 0.f);
        float tv = sg * (float)(NSTEPS - s);
        if (tv < bestval) {
          bestval = tv; ind = s; z_ind = z; s_ind = sv;
          zlo = zps; slo = sps;      // previous step = ind-1 (patched for ind==0)
        }
        zps = z; sps = sv;
      }
    }
  }
  if (ri < 16 && ind == 0) { zlo = zps; slo = sps; }  // (ind-1)%N wraps to step 99

  // ---- secant refinement
  bool net_surface = false;
  float zl = 0, sl = 0, zh = 0, sh = 0, zp = 0;
  if (ri < 16) {
    net_surface = (s_ind < 0.f) && sampler_mask;
    zl = zlo; sl = slo; zh = z_ind; sh = s_ind;
    float den = sh - sl; if (fabsf(den) <= 1e-12f) den = 1e-12f;
    zp = -sl * (zh - zl) / den + zl;
  }
  #pragma unroll 1
  for (int it = 0; it < NSECANT; ++it) {
    if (ri < 16) {
      pts[ri * 3 + 0] = cx + zp * rx; pts[ri * 3 + 1] = cy + zp * ry; pts[ri * 3 + 2] = cz + zp * rz;
      pts[(16 + ri) * 3 + 0] = cx; pts[(16 + ri) * 3 + 1] = cy; pts[(16 + ri) * 3 + 2] = cz; // pad slot
    }
    sdf_eval(smem, tid);
    if (ri < 16) {
      float sm = sdfv[ri];
      if (sm > 0.f) { zl = zp; sl = sm; }
      if (sm < 0.f) { zh = zp; sh = sm; }
      float den = sh - sl; if (fabsf(den) <= 1e-12f) den = 1e-12f;
      zp = -sl * (zh - zl) / den + zl;
    }
  }

  // ---- outputs: [points R*3 | net_obj R | dists R]
  if (ri < 16) {
    int g = r0 + ri;
    if (g < R) {
      float sd = net_surface ? zp : z_ind;
      float d  = sampler_mask ? sd : acc_s;
      bool  no = sampler_mask ? net_surface : net_obj;
      float tt = sampler_mask ? sd : acc_s;
      out[g * 3 + 0] = cx + tt * rx;
      out[g * 3 + 1] = cy + tt * ry;
      out[g * 3 + 2] = cz + tt * rz;
      out[R * 3 + g] = no ? 1.f : 0.f;
      out[R * 4 + g] = d;
    }
  }
}

extern "C" void kernel_launch(void* const* d_in, const int* in_sizes, int n_in,
                              void* d_out, int out_size, void* d_ws, size_t ws_size,
                              hipStream_t stream) {
  (void)n_in; (void)out_size; (void)d_ws; (void)ws_size;
  const float* cam = (const float*)d_in[0];
  const float* ray = (const float*)d_in[1];
  // d_in[2] = object_mask (unused in eval path)
  const float* W1 = (const float*)d_in[3];
  const float* b1 = (const float*)d_in[4];
  const float* W2 = (const float*)d_in[5];
  const float* b2 = (const float*)d_in[6];
  const float* W3 = (const float*)d_in[7];
  const float* b3 = (const float*)d_in[8];
  int R = in_sizes[0] / 3;
  int blocks = (R + 15) / 16;
  raytrace_kernel<<<blocks, 256, LDS_TOTAL, stream>>>(
      cam, ray, W1, b1, W2, b2, W3, b3, (float*)d_out, R);
}
